// MelDecoder_42399917146605
// MI455X (gfx1250) — compile-verified, matrix operand order checked by reading, not run on hardware
//
#include <hip/hip_runtime.h>
#include <hip/hip_fp16.h>
#include <math.h>

// ---------------------------------------------------------------------------
// DDSP vocoder for MI455X (gfx1250, wave32).
// GEMM-shaped work (MLP layers, irfft-as-cosine-basis) -> v_wmma_f32_16x16x32_f16
// Everything else -> simple wave32 VALU kernels; working set fits in 192MB L2.
// ---------------------------------------------------------------------------

typedef __attribute__((ext_vector_type(16))) _Float16 v16h;
typedef __attribute__((ext_vector_type(8)))  float    v8f;

#define SRATE     24000
#define HOP       240
#define NHARM     80
#define NB        4
#define NT        500
#define ROWS      (NB * NT)        // 2000 frames
#define NSAMP     (NT * HOP)       // 120000 samples / batch
#define FEAT_K    256              // 233 real features, zero-padded to 256
#define HID       256
#define NMAGH     256
#define NMAGN     80
#define NMAGN_PAD 96               // pad K for WMMA (multiple of 32)
#define CTRL_N    336              // 256 + 80
#define IRH       510              // harmonic ir_size = 2*(256-1)
#define IRH_PAD   512
#define IRN       158              // noise ir_size = 2*(80-1)
#define IRN_PAD   160
#define TWO_PI    6.28318530717958647692f

// ---------------------------------------------------------------------------
// WMMA helpers (wave32). A is row-major [M,K] f16; B is pre-swizzled by the
// prep kernels into the per-lane fragment order (K-contiguous per output
// column), so both operands use the same 2x b128 per-lane load pattern.
// 16-bit A fragment: lanes 0-15 row M=lane, K in {k0..k0+7, k0+16..k0+23};
// lanes 16-31 same row, K in {k0+8..k0+15, k0+24..k0+31}.
// ---------------------------------------------------------------------------
__device__ __forceinline__ v16h ld_frag(const _Float16* base, int ld,
                                        int row0, int k0, int lane) {
  const _Float16* p = base + (size_t)(row0 + (lane & 15)) * ld + k0 + ((lane >> 4) & 1) * 8;
  union { uint4 q[2]; v16h v; } fr;
  fr.q[0] = *reinterpret_cast<const uint4*>(p);        // K .. K+7   (16B)
  fr.q[1] = *reinterpret_cast<const uint4*>(p + 16);   // K+16..K+23 (16B)
  return fr.v;
}

__device__ __forceinline__ v8f wmma_tile(const _Float16* A, const _Float16* Bt,
                                         int K, int m0, int n0, int lane) {
  v8f acc = {0.f, 0.f, 0.f, 0.f, 0.f, 0.f, 0.f, 0.f};
  for (int k = 0; k < K; k += 32) {
    v16h a = ld_frag(A,  K, m0, k, lane);
    v16h b = ld_frag(Bt, K, n0, k, lane);
    acc = __builtin_amdgcn_wmma_f32_16x16x32_f16(false, a, false, b,
                                                 (short)0, acc, false, false);
  }
  return acc;
}

// C/D fragment layout: VGPR v -> row m0 + v + 8*(lane>=16), col n0 + (lane&15)

// ---------------------------------------------------------------------------
// Prep kernels: weight/basis transposition + f16 conversion.
// ---------------------------------------------------------------------------
__global__ void k_w1t(const float* __restrict__ W1, _Float16* __restrict__ W1t) {
  int i = blockIdx.x * blockDim.x + threadIdx.x;          // 256*256
  if (i >= 256 * 256) return;
  int n = i >> 8, k = i & 255;
  float v = (k < 233) ? W1[k * 256 + n] : 0.f;            // W1: [233,256]
  W1t[n * 256 + k] = (_Float16)v;
}

__global__ void k_w2t(const float* __restrict__ W2, _Float16* __restrict__ W2t) {
  int i = blockIdx.x * blockDim.x + threadIdx.x;          // 336*256
  if (i >= CTRL_N * 256) return;
  int n = i / 256, k = i % 256;                           // W2: [256,336]
  W2t[i] = (_Float16)W2[k * CTRL_N + n];
}

// Cosine basis with roll(ir_size/2) + Hann window folded in:
// ir_causal[n] = win[n] * (1/N) * sum_k ck * mag[k] * cos(2*pi*k*(n - N/2)/N)
__global__ void k_ch(_Float16* __restrict__ Ch) {
  int i = blockIdx.x * blockDim.x + threadIdx.x;          // 512*256
  if (i >= IRH_PAD * 256) return;
  int n = i >> 8, k = i & 255;
  float v = 0.f;
  if (n < IRH) {
    float win = 0.5f * (1.f - cosf(TWO_PI * (float)n / (float)(IRH - 1)));
    float ck  = (k == 0 || k == 255) ? 1.f : 2.f;
    v = win * ck * cosf(TWO_PI * (float)k * (float)(n - IRH / 2) / (float)IRH) * (1.f / (float)IRH);
  }
  Ch[i] = (_Float16)v;
}

__global__ void k_cn(_Float16* __restrict__ Cn) {
  int i = blockIdx.x * blockDim.x + threadIdx.x;          // 160*96
  if (i >= IRN_PAD * NMAGN_PAD) return;
  int n = i / NMAGN_PAD, k = i % NMAGN_PAD;
  float v = 0.f;
  if (n < IRN && k < NMAGN) {
    float win = 0.5f * (1.f - cosf(TWO_PI * (float)n / (float)(IRN - 1)));
    float ck  = (k == 0 || k == 79) ? 1.f : 2.f;
    v = win * ck * cosf(TWO_PI * (float)k * (float)(n - IRN / 2) / (float)IRN) * (1.f / (float)IRN);
  }
  Cn[i] = (_Float16)v;
}

__global__ void k_zero_noisepad(_Float16* __restrict__ noiseM) {
  int i = blockIdx.x * blockDim.x + threadIdx.x;          // 2000*16 pad cols
  if (i >= ROWS * 16) return;
  noiseM[(i / 16) * NMAGN_PAD + NMAGN + (i % 16)] = (_Float16)0.f;
}

// ---------------------------------------------------------------------------
// Feature assembly: [mel(80) | f0(1) | phoneme(128) | singer(16) | lang(8) | 0-pad]
// ---------------------------------------------------------------------------
__global__ void k_feats(const float* __restrict__ mel, const float* __restrict__ f0,
                        const int* __restrict__ phon, const int* __restrict__ sid,
                        const int* __restrict__ lid, const float* __restrict__ pht,
                        const float* __restrict__ sgt, const float* __restrict__ lgt,
                        _Float16* __restrict__ feats) {
  int i = blockIdx.x * blockDim.x + threadIdx.x;          // 2000*256
  if (i >= ROWS * FEAT_K) return;
  int r = i >> 8, c = i & 255;
  int b = r / NT;
  float v;
  if      (c < 80)  v = mel[r * 80 + c];
  else if (c == 80) v = f0[r];
  else if (c < 209) v = pht[phon[r] * 128 + (c - 81)];
  else if (c < 225) v = sgt[sid[b] * 16 + (c - 209)];
  else if (c < 233) v = lgt[lid[b] * 8 + (c - 225)];
  else              v = 0.f;
  feats[i] = (_Float16)v;
}

// ---------------------------------------------------------------------------
// WMMA GEMM kernels (8 waves / block, one 16x16 tile per wave).
// ---------------------------------------------------------------------------
__global__ void k_gemm1(const _Float16* __restrict__ feats, const _Float16* __restrict__ W1t,
                        const float* __restrict__ b1, _Float16* __restrict__ h) {
  int wave = (blockIdx.x * blockDim.x + threadIdx.x) >> 5;
  int lane = threadIdx.x & 31;
  const int tilesN = HID / 16, tiles = (ROWS / 16) * tilesN;
  if (wave >= tiles) return;
  int m0 = (wave / tilesN) * 16, n0 = (wave % tilesN) * 16;
  v8f acc = wmma_tile(feats, W1t, FEAT_K, m0, n0, lane);
  int col = n0 + (lane & 15);
  int rb  = m0 + ((lane >> 4) & 1) * 8;
  float bias = b1[col];
#pragma unroll
  for (int v = 0; v < 8; ++v) {
    float x = acc[v] + bias;
    h[(size_t)(rb + v) * HID + col] = (_Float16)fmaxf(x, 0.f);   // ReLU
  }
}

__global__ void k_gemm2(const _Float16* __restrict__ h, const _Float16* __restrict__ W2t,
                        const float* __restrict__ b2, _Float16* __restrict__ harm,
                        _Float16* __restrict__ noiseM) {
  int wave = (blockIdx.x * blockDim.x + threadIdx.x) >> 5;
  int lane = threadIdx.x & 31;
  const int tilesN = CTRL_N / 16, tiles = (ROWS / 16) * tilesN;  // 125*21
  if (wave >= tiles) return;
  int m0 = (wave / tilesN) * 16, n0 = (wave % tilesN) * 16;
  v8f acc = wmma_tile(h, W2t, HID, m0, n0, lane);
  int col = n0 + (lane & 15);
  int rb  = m0 + ((lane >> 4) & 1) * 8;
  float bias = b2[col];
#pragma unroll
  for (int v = 0; v < 8; ++v) {
    float x  = acc[v] + bias;
    float sg = 1.f / (1.f + __expf(-x));
    // scale_fn: 2*sigmoid(x)^ln(10) + 1e-7
    float y  = 2.f * __expf(2.30258509299404568402f * __logf(sg)) + 1e-7f;
    int row = rb + v;
    if (col < NMAGH) harm[(size_t)row * NMAGH + col] = (_Float16)y;
    else             noiseM[(size_t)row * NMAGN_PAD + (col - NMAGH)] = (_Float16)y;
  }
}

// irfft-as-GEMM: out[r][n] = sum_k mag[r][k] * Cbasis[n][k]  (causal windowed IR)
__global__ void k_gemm_ir(const _Float16* __restrict__ A, const _Float16* __restrict__ Bt,
                          float* __restrict__ out, int K, int N, int tilesN, int tiles) {
  int wave = (blockIdx.x * blockDim.x + threadIdx.x) >> 5;
  int lane = threadIdx.x & 31;
  if (wave >= tiles) return;
  int m0 = (wave / tilesN) * 16, n0 = (wave % tilesN) * 16;
  v8f acc = wmma_tile(A, Bt, K, m0, n0, lane);
  int col = n0 + (lane & 15);
  int rb  = m0 + ((lane >> 4) & 1) * 8;
#pragma unroll
  for (int v = 0; v < 8; ++v)
    out[(size_t)(rb + v) * N + col] = acc[v];
}

// ---------------------------------------------------------------------------
// Pitch: clamp/threshold per frame, then analytic scan of the piecewise-linear
// upsampled cumsum. Phi is stored mod 2*pi (exact for integer harmonics).
// frame_sum = 240*p0 + (p1-p0)*119.5
// ---------------------------------------------------------------------------
__global__ void k_pitch(const float* __restrict__ f0, float* __restrict__ pbuf,
                        float* __restrict__ Phi) {
  int b = blockIdx.x;
  for (int t = threadIdx.x; t < NT; t += blockDim.x) {
    float p = f0[b * NT + t];
    p = fminf(fmaxf(p, 0.f), 1000.f);
    if (p < 80.f) p = 0.f;
    pbuf[b * (NT + 1) + t] = p;
  }
  __syncthreads();
  if (threadIdx.x == 0) {
    pbuf[b * (NT + 1) + NT] = pbuf[b * (NT + 1) + NT - 1];
    float acc = 0.f;
    for (int f = 0; f < NT; ++f) {
      Phi[b * NT + f] = fmodf(acc * (TWO_PI / (float)SRATE), TWO_PI);
      float p0 = pbuf[b * (NT + 1) + f], p1 = pbuf[b * (NT + 1) + f + 1];
      acc += 240.f * p0 + (p1 - p0) * 119.5f;
    }
  }
}

// 80-harmonic oscillator. Phase in-frame (inclusive cumsum) is closed-form:
// phase = Phi_f + (2pi/SR)*((i+1)*p0 + dp*i*(i+1)/(2*HOP))
__global__ void k_osc(const float* __restrict__ pbuf, const float* __restrict__ Phi,
                      float* __restrict__ xh) {
  int idx = blockIdx.x * blockDim.x + threadIdx.x;
  if (idx >= NB * NSAMP) return;
  int b = idx / NSAMP, n = idx % NSAMP;
  int f = n / HOP, i = n - f * HOP;
  float p0 = pbuf[b * (NT + 1) + f], p1 = pbuf[b * (NT + 1) + f + 1];
  float dp = p1 - p0, fi = (float)i;
  float pit = p0 + dp * fi * (1.f / (float)HOP);
  float ph  = Phi[b * NT + f] +
              (TWO_PI / (float)SRATE) * ((fi + 1.f) * p0 + dp * fi * (fi + 1.f) * (1.f / (2.f * HOP)));
  ph = fmodf(ph, TWO_PI);            // sin(k*(ph mod 2pi)) == sin(k*ph), k integer
  float s = 0.f;
#pragma unroll 8
  for (int k = 1; k <= NHARM; ++k) {
    float fk = (float)k;
    if (pit * fk < 12000.f)          // Nyquist mask
      s += __sinf(ph * fk) * (1.f / fk);
  }
  xh[idx] = 0.4f * s;
}

// ---------------------------------------------------------------------------
// Time-varying convolution (exact equivalent of per-frame FFT conv + OLA +
// group-delay trim): out[n] = sum_tau x[n+d-tau] * ir[frame(n+d-tau)][tau]
// Entire x/ir working set lives in the 192MB L2.
// ---------------------------------------------------------------------------
__global__ void k_conv(const float* __restrict__ xh, const float* __restrict__ noise,
                       const float* __restrict__ irh, const float* __restrict__ irn,
                       float* __restrict__ out) {
  int idx = blockIdx.x * blockDim.x + threadIdx.x;
  if (idx >= NB * NSAMP) return;
  int b = idx / NSAMP, n = idx % NSAMP;
  float acc = 0.f;
  {
    const int d = IRH / 2;                                   // 255
    int smin = max(n - (IRH - 1 - d), 0), smax = min(n + d, NSAMP - 1);
    int f = smin / HOP, rem = smin - f * HOP;
    const float* xb  = xh  + (size_t)b * NSAMP;
    const float* irb = irh + (size_t)b * NT * IRH_PAD;
    for (int s = smin; s <= smax; ++s) {
      acc += xb[s] * irb[f * IRH_PAD + (n + d - s)];
      if (++rem == HOP) { rem = 0; ++f; }
    }
  }
  {
    const int d = IRN / 2;                                   // 79
    int smin = max(n - (IRN - 1 - d), 0), smax = min(n + d, NSAMP - 1);
    int f = smin / HOP, rem = smin - f * HOP;
    const float* nb  = noise + (size_t)b * NSAMP;
    const float* irb = irn   + (size_t)b * NT * IRN_PAD;
    for (int s = smin; s <= smax; ++s) {
      acc += (2.f * nb[s] - 1.f) * irb[f * IRN_PAD + (n + d - s)];
      if (++rem == HOP) { rem = 0; ++f; }
    }
  }
  out[idx] = acc;
}

// ---------------------------------------------------------------------------
extern "C" void kernel_launch(void* const* d_in, const int* in_sizes, int n_in,
                              void* d_out, int out_size, void* d_ws, size_t ws_size,
                              hipStream_t stream) {
  (void)in_sizes; (void)n_in; (void)out_size; (void)ws_size;
  const float* mel   = (const float*)d_in[0];
  const float* f0    = (const float*)d_in[1];
  const int*   phon  = (const int*)d_in[2];
  const int*   sid   = (const int*)d_in[3];
  const int*   lid   = (const int*)d_in[4];
  const float* noise = (const float*)d_in[5];
  const float* pht   = (const float*)d_in[6];
  const float* sgt   = (const float*)d_in[7];
  const float* lgt   = (const float*)d_in[8];
  const float* W1    = (const float*)d_in[9];
  const float* b1    = (const float*)d_in[10];
  const float* W2    = (const float*)d_in[11];
  const float* b2    = (const float*)d_in[12];
  float* out = (float*)d_out;

  // workspace carve-out (256B aligned slices)
  char* w = (char*)d_ws;
  auto carve = [&](size_t bytes) { char* p = w; w += (bytes + 255) & ~(size_t)255; return p; };
  _Float16* feats  = (_Float16*)carve((size_t)ROWS * FEAT_K * 2);
  _Float16* h      = (_Float16*)carve((size_t)ROWS * HID * 2);
  _Float16* W1t    = (_Float16*)carve((size_t)256 * 256 * 2);
  _Float16* W2t    = (_Float16*)carve((size_t)CTRL_N * 256 * 2);
  _Float16* harm   = (_Float16*)carve((size_t)ROWS * NMAGH * 2);
  _Float16* noiseM = (_Float16*)carve((size_t)ROWS * NMAGN_PAD * 2);
  _Float16* Ch     = (_Float16*)carve((size_t)IRH_PAD * 256 * 2);
  _Float16* Cn     = (_Float16*)carve((size_t)IRN_PAD * NMAGN_PAD * 2);
  float*    irh    = (float*)carve((size_t)ROWS * IRH_PAD * 4);
  float*    irn    = (float*)carve((size_t)ROWS * IRN_PAD * 4);
  float*    pbuf   = (float*)carve((size_t)NB * (NT + 1) * 4);
  float*    Phi    = (float*)carve((size_t)NB * NT * 4);
  float*    xh     = (float*)carve((size_t)NB * NSAMP * 4);

  const int BT = 256;
  // constants / weights / basis
  k_w1t<<<(256 * 256 + BT - 1) / BT, BT, 0, stream>>>(W1, W1t);
  k_w2t<<<(CTRL_N * 256 + BT - 1) / BT, BT, 0, stream>>>(W2, W2t);
  k_ch <<<(IRH_PAD * 256 + BT - 1) / BT, BT, 0, stream>>>(Ch);
  k_cn <<<(IRN_PAD * NMAGN_PAD + BT - 1) / BT, BT, 0, stream>>>(Cn);
  k_zero_noisepad<<<(ROWS * 16 + BT - 1) / BT, BT, 0, stream>>>(noiseM);
  // features
  k_feats<<<(ROWS * FEAT_K + BT - 1) / BT, BT, 0, stream>>>(
      mel, f0, phon, sid, lid, pht, sgt, lgt, feats);
  // MLP (WMMA)
  {
    int tiles = (ROWS / 16) * (HID / 16);                    // 2000
    k_gemm1<<<(tiles + 7) / 8, BT, 0, stream>>>(feats, W1t, b1, h);
  }
  {
    int tiles = (ROWS / 16) * (CTRL_N / 16);                 // 2625
    k_gemm2<<<(tiles + 7) / 8, BT, 0, stream>>>(h, W2t, b2, harm, noiseM);
  }
  // irfft-as-GEMM (WMMA)
  {
    int tilesN = IRH_PAD / 16, tiles = (ROWS / 16) * tilesN; // 4000
    k_gemm_ir<<<(tiles + 7) / 8, BT, 0, stream>>>(harm, Ch, irh, NMAGH, IRH_PAD, tilesN, tiles);
  }
  {
    int tilesN = IRN_PAD / 16, tiles = (ROWS / 16) * tilesN; // 1250
    k_gemm_ir<<<(tiles + 7) / 8, BT, 0, stream>>>(noiseM, Cn, irn, NMAGN_PAD, IRN_PAD, tilesN, tiles);
  }
  // pitch scan + oscillator + time-varying convolution
  k_pitch<<<NB, 32, 0, stream>>>(f0, pbuf, Phi);
  k_osc  <<<(NB * NSAMP + BT - 1) / BT, BT, 0, stream>>>(pbuf, Phi, xh);
  k_conv <<<(NB * NSAMP + BT - 1) / BT, BT, 0, stream>>>(xh, noise, irh, irn, out);
}
